// SpectralMoleculeEncoder_38062000177587
// MI455X (gfx1250) — matrix operand order repaired and checked by reading, not compiled, not run on hardware
//
#include <hip/hip_runtime.h>
#include <hip/hip_bf16.h>

typedef __bf16 bf16_t;
typedef __attribute__((ext_vector_type(16))) __bf16 v16bf;
typedef __attribute__((ext_vector_type(8)))  __bf16 v8bf;
typedef __attribute__((ext_vector_type(8)))  float  v8f;

#define TB 256

// ---------------- utility kernels ----------------

__global__ void k_zero_f32(float* __restrict__ p, long long n) {
    long long i = (long long)blockIdx.x * blockDim.x + threadIdx.x;
    if (i < n) p[i] = 0.0f;
}

__global__ void k_degree(const long long* __restrict__ src, float* __restrict__ deg, int E) {
    int e = blockIdx.x * blockDim.x + threadIdx.x;
    if (e < E) atomicAdd(&deg[(int)src[e]], 1.0f);
}

__global__ void k_dinv(float* __restrict__ deg, int n) {
    int i = blockIdx.x * blockDim.x + threadIdx.x;
    if (i < n) {
        float d = deg[i];
        deg[i] = (d > 0.0f) ? rsqrtf(fmaxf(d, 1.0f)) : 0.0f;
    }
}

__global__ void k_norm(const long long* __restrict__ src, const long long* __restrict__ dst,
                       const float* __restrict__ dinv, float* __restrict__ norm, int E) {
    int e = blockIdx.x * blockDim.x + threadIdx.x;
    if (e < E) norm[e] = -dinv[(int)src[e]] * dinv[(int)dst[e]];
}

// out[dst, :] += norm_e * x[src, :]   (4 channels per thread, f32 atomics into L2)
__global__ void k_prop_scatter(const float* __restrict__ x,
                               const long long* __restrict__ src,
                               const long long* __restrict__ dst,
                               const float* __restrict__ norm,
                               float* __restrict__ out, int E, int F) {
    long long tid = (long long)blockIdx.x * blockDim.x + threadIdx.x;
    int chunks = F >> 2;
    long long total = (long long)E * chunks;
    if (tid >= total) return;
    int e = (int)(tid / chunks);
    int c = (int)(tid % chunks) << 2;
    int s = (int)src[e], d = (int)dst[e];
    float w = norm[e];
    const float4 xv = *(const float4*)(x + (size_t)s * F + c);
    float* o = out + (size_t)d * F + c;
    atomicAdd(o + 0, w * xv.x);
    atomicAdd(o + 1, w * xv.y);
    atomicAdd(o + 2, w * xv.z);
    atomicAdd(o + 3, w * xv.w);
}

// t2 = 2*t2 - x  (Chebyshev recurrence Tx2)
__global__ void k_tx2(float* __restrict__ t2, const float* __restrict__ x, long long n) {
    long long i = (long long)blockIdx.x * blockDim.x + threadIdx.x;
    if (i < n) t2[i] = 2.0f * t2[i] - x[i];
}

// ---------------- fragment repack kernels ----------------
// Fragment-major storage: frag[tile][kstep][lane][i] with 16 contiguous bf16
// per lane -> two global_load_b128 per fragment in the GEMM, fully coalesced.

// A-operand (ISA 16-bit A 16x32 layout):
//   lanes 0-15 : row = lane,      K = {0..7} -> i 0..7, {16..23} -> i 8..15
//   lanes 16-31: row = lane-16,   K = {8..15}-> i 0..7, {24..31} -> i 8..15
__global__ void k_pack3_afrag(const float* __restrict__ x0, const float* __restrict__ x1,
                              const float* __restrict__ x2, bf16_t* __restrict__ afrag,
                              int N, int F) {
    long long i = (long long)blockIdx.x * blockDim.x + threadIdx.x;
    int K = 3 * F;
    long long n = (long long)N * K;
    if (i >= n) return;
    int row  = (int)(i / K);
    int col3 = (int)(i % K);
    const float* s = (col3 < F) ? x0 : (col3 < 2 * F) ? x1 : x2;
    int col = (col3 < F) ? col3 : (col3 < 2 * F) ? col3 - F : col3 - 2 * F;
    float val = s[(size_t)row * F + col];

    int KS = K >> 5;
    int mt = row >> 4, rr = row & 15;
    int ks = col3 >> 5, cc = col3 & 31;
    int chunk = cc >> 3;                          // 0..3
    int lane = rr + ((chunk & 1) ? 16 : 0);
    int idx  = (cc & 7) + ((chunk >> 1) ? 8 : 0);
    afrag[(((size_t)mt * KS + ks) * 32 + lane) * 16 + idx] = (bf16_t)val;
}

// B-operand (ISA 16-bit B 32x16 layout):
//   lanes 0-15 : col = lane,    K = {0..7} -> i 0..7, {16..23} -> i 8..15
//   lanes 16-31: col = lane-16, K = {8..15}-> i 0..7, {24..31} -> i 8..15
// W[3,Fin,Fout] flat is already [K, Nc] row-major.
__global__ void k_pack_bfrag(const float* __restrict__ Wf, bf16_t* __restrict__ bfrag,
                             int K, int Nc) {
    long long i = (long long)blockIdx.x * blockDim.x + threadIdx.x;
    long long n = (long long)K * Nc;
    if (i >= n) return;
    int k = (int)(i / Nc);
    int c = (int)(i % Nc);
    float val = Wf[i];

    int KS = K >> 5;
    int ks = k >> 5, kc = k & 31;
    int nt16 = c >> 4, cc = c & 15;
    int lane = cc + (((kc >> 3) & 1) ? 16 : 0);
    int idx  = (kc & 7) + ((kc >> 4) ? 8 : 0);
    bfrag[(((size_t)nt16 * KS + ks) * 32 + lane) * 16 + idx] = (bf16_t)val;
}

// ---------------- WMMA GEMM: Y = relu(A @ B + bias) ----------------
// A: fragment-major [Mtiles(+1 pad)][KS][32][16] bf16, B: [Ntiles16][KS][32][16].
// One wave -> 32(M) x 64(N) output tile: 8 accumulators; per K-step:
//   4 b128 A-loads + 8 b128 B-loads + 8 v_wmma_f32_16x16x32_bf16 (1.5 loads/WMMA).
// __launch_bounds__(256, 1): lift VGPR cap so the 64 accumulator VGPRs stay
// resident (round-3 build spilled them to scratch under the default budget).
template <int KS>
__global__ void __launch_bounds__(256, 1)
k_wmma_gemm_bias_relu(const bf16_t* __restrict__ A,
                      const bf16_t* __restrict__ B,
                      const float*  __restrict__ bias,
                      float* __restrict__ Y,
                      int Mtiles, int Nc) {
    const int lane = threadIdx.x & 31;
    const int wavesPerBlk = blockDim.x >> 5;
    const int ntn = Nc >> 6;                     // 64-wide N supertiles
    const int mts = (Mtiles + 1) >> 1;           // 32-row M supertiles
    const long long wid = (long long)blockIdx.x * wavesPerBlk + (threadIdx.x >> 5);
    const long long total = (long long)mts * ntn;
    if (wid >= total) return;                    // wave-uniform

    const int mt2 = (int)(wid / ntn);
    const int nt  = (int)(wid % ntn);
    const int mt0 = mt2 * 2;
    const int mt1 = mt0 + 1;

    const bf16_t* ab0 = A + ((size_t)mt0 * KS) * 512 + lane * 16;   // 512 = 32 lanes * 16
    const bf16_t* ab1 = ab0 + (size_t)KS * 512;
    const bf16_t* bb  = B + ((size_t)(nt * 4) * KS) * 512 + lane * 16;

    v8f c00 = {}, c01 = {}, c02 = {}, c03 = {};
    v8f c10 = {}, c11 = {}, c12 = {}, c13 = {};

    for (int ks = 0; ks < KS; ++ks) {
        const bf16_t* a0p = ab0 + (size_t)ks * 512;
        const bf16_t* a1p = ab1 + (size_t)ks * 512;
        v16bf a0 = __builtin_shufflevector(*(const v8bf*)a0p, *(const v8bf*)(a0p + 8),
                                           0, 1, 2, 3, 4, 5, 6, 7, 8, 9, 10, 11, 12, 13, 14, 15);
        v16bf a1 = __builtin_shufflevector(*(const v8bf*)a1p, *(const v8bf*)(a1p + 8),
                                           0, 1, 2, 3, 4, 5, 6, 7, 8, 9, 10, 11, 12, 13, 14, 15);

        const bf16_t* bp0 = bb + ((size_t)(0 * KS + ks)) * 512;
        const bf16_t* bp1 = bb + ((size_t)(1 * KS + ks)) * 512;
        const bf16_t* bp2 = bb + ((size_t)(2 * KS + ks)) * 512;
        const bf16_t* bp3 = bb + ((size_t)(3 * KS + ks)) * 512;
        v16bf b0 = __builtin_shufflevector(*(const v8bf*)bp0, *(const v8bf*)(bp0 + 8),
                                           0, 1, 2, 3, 4, 5, 6, 7, 8, 9, 10, 11, 12, 13, 14, 15);
        v16bf b1 = __builtin_shufflevector(*(const v8bf*)bp1, *(const v8bf*)(bp1 + 8),
                                           0, 1, 2, 3, 4, 5, 6, 7, 8, 9, 10, 11, 12, 13, 14, 15);
        v16bf b2 = __builtin_shufflevector(*(const v8bf*)bp2, *(const v8bf*)(bp2 + 8),
                                           0, 1, 2, 3, 4, 5, 6, 7, 8, 9, 10, 11, 12, 13, 14, 15);
        v16bf b3 = __builtin_shufflevector(*(const v8bf*)bp3, *(const v8bf*)(bp3 + 8),
                                           0, 1, 2, 3, 4, 5, 6, 7, 8, 9, 10, 11, 12, 13, 14, 15);

        c00 = __builtin_amdgcn_wmma_f32_16x16x32_bf16(false, a0, false, b0, (short)0, c00, false, false);
        c01 = __builtin_amdgcn_wmma_f32_16x16x32_bf16(false, a0, false, b1, (short)0, c01, false, false);
        c02 = __builtin_amdgcn_wmma_f32_16x16x32_bf16(false, a0, false, b2, (short)0, c02, false, false);
        c03 = __builtin_amdgcn_wmma_f32_16x16x32_bf16(false, a0, false, b3, (short)0, c03, false, false);
        c10 = __builtin_amdgcn_wmma_f32_16x16x32_bf16(false, a1, false, b0, (short)0, c10, false, false);
        c11 = __builtin_amdgcn_wmma_f32_16x16x32_bf16(false, a1, false, b1, (short)0, c11, false, false);
        c12 = __builtin_amdgcn_wmma_f32_16x16x32_bf16(false, a1, false, b2, (short)0, c12, false, false);
        c13 = __builtin_amdgcn_wmma_f32_16x16x32_bf16(false, a1, false, b3, (short)0, c13, false, false);
    }

    // D layout: VGPR r -> M = r (lanes 0-15) / 8+r (lanes 16-31); lane%16 -> N
    const int hi = lane >> 4;
    const int col0 = nt * 64 + (lane & 15);

    {   // M-tile 0 (always valid)
        const int rowbase = mt0 * 16 + (hi ? 8 : 0);
#pragma unroll
        for (int j = 0; j < 4; ++j) {
            v8f c = (j == 0) ? c00 : (j == 1) ? c01 : (j == 2) ? c02 : c03;
            int col = col0 + j * 16;
            float bv = bias[col];
#pragma unroll
            for (int r = 0; r < 8; ++r) {
                float v = c[r] + bv;
                Y[(size_t)(rowbase + r) * Nc + col] = v > 0.0f ? v : 0.0f;
            }
        }
    }
    if (mt1 < Mtiles) {  // M-tile 1 (phantom on last supertile) — wave-uniform guard
        const int rowbase = mt1 * 16 + (hi ? 8 : 0);
#pragma unroll
        for (int j = 0; j < 4; ++j) {
            v8f c = (j == 0) ? c10 : (j == 1) ? c11 : (j == 2) ? c12 : c13;
            int col = col0 + j * 16;
            float bv = bias[col];
#pragma unroll
            for (int r = 0; r < 8; ++r) {
                float v = c[r] + bv;
                Y[(size_t)(rowbase + r) * Nc + col] = v > 0.0f ? v : 0.0f;
            }
        }
    }
}

// ---------------- host side ----------------

static inline unsigned nblk(long long n, int tb) { return (unsigned)((n + tb - 1) / tb); }

extern "C" void kernel_launch(void* const* d_in, const int* in_sizes, int n_in,
                              void* d_out, int out_size, void* d_ws, size_t ws_size,
                              hipStream_t stream) {
    const float*     v   = (const float*)d_in[0];
    const long long* ei  = (const long long*)d_in[1];   // int64 [2, E]
    const float*     W1  = (const float*)d_in[2];
    const float*     b1  = (const float*)d_in[3];
    const float*     W2  = (const float*)d_in[4];
    const float*     b2  = (const float*)d_in[5];
    const float*     W3  = (const float*)d_in[6];
    const float*     b3  = (const float*)d_in[7];
    float*           out = (float*)d_out;

    const int N = 50000;
    const int E = in_sizes[1] / 2;
    const long long* src = ei;
    const long long* dst = ei + E;

    // workspace carve-up (256B aligned)
    char* p = (char*)d_ws;
    auto carve = [&](size_t bytes) -> char* {
        char* r = p;
        p += (bytes + 255) & ~(size_t)255;
        return r;
    };
    float*  deg   = (float*)carve((size_t)N * 4);             // becomes dinv in place
    float*  norm  = (float*)carve((size_t)E * 4);
    float*  t1    = (float*)carve((size_t)N * 256 * 4);
    float*  t2    = (float*)carve((size_t)N * 256 * 4);
    bf16_t* afrag = (bf16_t*)carve((size_t)N * 768 * 2 + 32768);  // +1 phantom tile pad
    bf16_t* bfrag = (bf16_t*)carve((size_t)768 * 512 * 2);
    float*  x1    = (float*)carve((size_t)N * 128 * 4);
    float*  x2    = (float*)carve((size_t)N * 256 * 4);

    // ---- graph normalization: deg -> dinv -> edge norm ----
    k_zero_f32<<<nblk(N, TB), TB, 0, stream>>>(deg, N);
    k_degree<<<nblk(E, TB), TB, 0, stream>>>(src, deg, E);
    k_dinv<<<nblk(N, TB), TB, 0, stream>>>(deg, N);
    k_norm<<<nblk(E, TB), TB, 0, stream>>>(src, dst, deg, norm, E);

    // ---- one ChebConv(K=3) + ReLU layer ----
    auto layer = [&](const float* xin, int Fin, int Fout,
                     const float* Wf, const float* bf, float* yout) {
        long long nf = (long long)N * Fin;
        long long sc = (long long)E * (Fin >> 2);

        k_zero_f32<<<nblk(nf, TB), TB, 0, stream>>>(t1, nf);
        k_prop_scatter<<<nblk(sc, TB), TB, 0, stream>>>(xin, src, dst, norm, t1, E, Fin);

        k_zero_f32<<<nblk(nf, TB), TB, 0, stream>>>(t2, nf);
        k_prop_scatter<<<nblk(sc, TB), TB, 0, stream>>>(t1, src, dst, norm, t2, E, Fin);

        k_tx2<<<nblk(nf, TB), TB, 0, stream>>>(t2, xin, nf);

        // operand repack into WMMA fragment-major layouts (bf16)
        k_pack3_afrag<<<nblk(3 * nf, TB), TB, 0, stream>>>(xin, t1, t2, afrag, N, Fin);
        int K = 3 * Fin;                                   // 96 / 384 / 768
        k_pack_bfrag<<<nblk((long long)K * Fout, TB), TB, 0, stream>>>(Wf, bfrag, K, Fout);

        int Mtiles = N / 16;                               // 3125
        long long tiles = (long long)((Mtiles + 1) / 2) * (Fout / 64);
        unsigned gblocks = (unsigned)((tiles + 7) / 8);    // 8 waves / 256-thread block
        if (K == 96)
            k_wmma_gemm_bias_relu<3><<<gblocks, 256, 0, stream>>>(afrag, bfrag, bf, yout, Mtiles, Fout);
        else if (K == 384)
            k_wmma_gemm_bias_relu<12><<<gblocks, 256, 0, stream>>>(afrag, bfrag, bf, yout, Mtiles, Fout);
        else
            k_wmma_gemm_bias_relu<24><<<gblocks, 256, 0, stream>>>(afrag, bfrag, bf, yout, Mtiles, Fout);
    };

    layer(v,  32, 128, W1, b1, x1);
    layer(x1, 128, 256, W2, b2, x2);
    layer(x2, 256, 512, W3, b3, out);
}
